// EdgeMessageLayer_65257733095556
// MI455X (gfx1250) — matrix operand order repaired and checked by reading, not compiled, not run on hardware
//
#include <hip/hip_runtime.h>
#include <hip/hip_bf16.h>
#include <stdint.h>

#define N_NODESC 100000
#define N_EDGESC 800000
#define HID 128
#define LN_EPSF 1e-5f

typedef __bf16 bf16_t;
typedef bf16_t v16bf __attribute__((ext_vector_type(16)));
typedef float  v8f   __attribute__((ext_vector_type(8)));

union BFrag { unsigned int u[8]; v16bf v; };

__device__ __forceinline__ unsigned int pack2bf(float a, float b) {
    unsigned int ua = __float_as_uint(a);
    unsigned int ub = __float_as_uint(b);
    ua = (ua + 0x7FFFu + ((ua >> 16) & 1u)) >> 16;   // RNE f32->bf16
    ub = (ub + 0x7FFFu + ((ub >> 16) & 1u)) >> 16;
    return (ua & 0xFFFFu) | (ub << 16);
}
__device__ __forceinline__ unsigned short bf1(float a) {
    unsigned int ua = __float_as_uint(a);
    return (unsigned short)((ua + 0x7FFFu + ((ua >> 16) & 1u)) >> 16);
}

// ---------------------------------------------------------------- zero scratch
__global__ void zero_kernel(float* __restrict__ p, size_t n) {
    size_t i = (size_t)blockIdx.x * blockDim.x + threadIdx.x;
    size_t stride = (size_t)gridDim.x * blockDim.x;
    for (; i < n; i += stride) p[i] = 0.0f;
}

// ------------------------------------------------- weight -> bf16 B-fragment swizzle
// Per 32x16 (K x N) tile: dword index = v*32 + lane packs W[K][n], W[K+1][n]
// with K = kt*32 + 16*(v>>2) + 8*(lane>>4) + 2*(v&3), n = nt*16 + (lane&15)
__global__ __launch_bounds__(256) void prep_kernel(
    const float* __restrict__ ep_w1, const float* __restrict__ ep_w2,
    const float* __restrict__ up_w1, const float* __restrict__ up_w2,
    unsigned int* __restrict__ w1e, unsigned int* __restrict__ w2e,
    unsigned int* __restrict__ w1u, unsigned int* __restrict__ w2u)
{
    int b = blockIdx.x;
    int tid = threadIdx.x;
    int v = tid >> 5, lane = tid & 31;
    const float* W; unsigned int* O; int kt, nt;
    if (b < 96)       { W = ep_w1; O = w1e; int r = b;        kt = r >> 3; nt = r & 7; }
    else if (b < 128) { W = ep_w2; O = w2e; int r = b - 96;   kt = r >> 3; nt = r & 7; }
    else if (b < 224) { W = up_w1; O = w1u; int r = b - 128;  kt = r >> 3; nt = r & 7; }
    else              { W = up_w2; O = w2u; int r = b - 224;  kt = r >> 3; nt = r & 7; }
    int K = kt * 32 + 16 * (v >> 2) + 8 * (lane >> 4) + 2 * (v & 3);
    int n = nt * 16 + (lane & 15);
    unsigned int val = pack2bf(W[(size_t)K * HID + n], W[(size_t)(K + 1) * HID + n]);
    O[(size_t)(kt * 8 + nt) * 256 + v * 32 + lane] = val;
}

// ---------------------------------------------------------------- edge phase
// 32 edges per tile, 8 waves x 16-col N-tiles, 2 M-subtiles per wave:
// B fragment loaded once per K-step feeds two back-to-back WMMAs.
__global__ __launch_bounds__(256) void edge_kernel(
    const float* __restrict__ h, const int* __restrict__ eidx,
    const float* __restrict__ ex, const float* __restrict__ b1,
    const float* __restrict__ b2, const unsigned int* __restrict__ w1g,
    const unsigned int* __restrict__ w2g, float* __restrict__ inc,
    float* __restrict__ outg, float* __restrict__ inDeg, float* __restrict__ outDeg)
{
    __shared__ unsigned int  w1s[12 * 8 * 256];   // 96 KB ep_w1 swizzled
    __shared__ unsigned int  w2s[4 * 8 * 256];    // 32 KB ep_w2 swizzled
    __shared__ unsigned int  aT[32 * 196];        // 32 x 384 bf16 (padded)
    __shared__ unsigned short hidH[32 * 132];     // 32 x 128 bf16 (padded)
    __shared__ int sSh[32], dSh[32];

    const int tid = threadIdx.x;
    for (int i = tid; i < 12 * 8 * 256; i += 256) w1s[i] = w1g[i];
    for (int i = tid; i < 4 * 8 * 256; i += 256)  w2s[i] = w2g[i];

    const int wave = tid >> 5, lane = tid & 31;
    const int half = lane >> 4, mrow = lane & 15, ncol = lane & 15;
    const int n0 = wave * 16;
    const float bias1 = b1[n0 + ncol];
    const float bias2 = b2[n0 + ncol];
    const unsigned int* hidU = (const unsigned int*)hidH;

    const int nTiles = N_EDGESC / 32;
    for (int t = blockIdx.x; t < nTiles; t += gridDim.x) {
        __syncthreads();   // protect sSh/dSh/aT reuse across iterations
        if (tid < 32) {
            int e = t * 32 + tid;
            int s = eidx[e];
            int d = eidx[N_EDGESC + e];
            s = s < 0 ? 0 : (s >= N_NODESC ? N_NODESC - 1 : s);
            d = d < 0 ? 0 : (d >= N_NODESC ? N_NODESC - 1 : d);
            sSh[tid] = s; dSh[tid] = d;
            atomicAdd(&outDeg[s], 1.0f);
            atomicAdd(&inDeg[d], 1.0f);
        }
        __syncthreads();
        // gather concat(h[src], h[dst], edge_x) -> bf16 tile (32 rows x 192 dword-pairs)
        for (int p = tid; p < 32 * 192; p += 256) {
            int m = p / 192, c = p - m * 192;
            int f0 = 2 * c;
            const float* sp;
            if (f0 < HID)            sp = h  + (size_t)sSh[m] * HID + f0;
            else if (f0 < 2 * HID)   sp = h  + (size_t)dSh[m] * HID + (f0 - HID);
            else                     sp = ex + (size_t)(t * 32 + m) * HID + (f0 - 2 * HID);
            aT[m * 196 + c] = pack2bf(sp[0], sp[1]);
        }
        if (tid == 0 && t + (int)gridDim.x < nTiles)
            __builtin_prefetch(ex + (size_t)(t + gridDim.x) * 32 * HID, 0, 1);
        __syncthreads();

        // GEMM1: hidden = relu([32x384] @ W1 + b1)
        v8f c0, c1;
        #pragma unroll
        for (int r = 0; r < 8; ++r) { c0[r] = bias1; c1[r] = bias1; }
        #pragma unroll
        for (int k = 0; k < 12; ++k) {
            BFrag a0, a1, b;
            #pragma unroll
            for (int v = 0; v < 8; ++v) {
                int sub = (v < 4) ? (4 * half + v) : (8 + 4 * half + (v - 4));
                b.u[v]  = w1s[(k * 8 + wave) * 256 + v * 32 + lane];
                a0.u[v] = aT[mrow * 196 + k * 16 + sub];
                a1.u[v] = aT[(16 + mrow) * 196 + k * 16 + sub];
            }
            c0 = __builtin_amdgcn_wmma_f32_16x16x32_bf16(false, a0.v, false, b.v,
                                                         (short)0, c0, false, false);
            c1 = __builtin_amdgcn_wmma_f32_16x16x32_bf16(false, a1.v, false, b.v,
                                                         (short)0, c1, false, false);
        }
        #pragma unroll
        for (int r = 0; r < 8; ++r) {
            int m = r + 8 * half;
            float v0 = c0[r] > 0.0f ? c0[r] : 0.0f;
            float v1 = c1[r] > 0.0f ? c1[r] : 0.0f;
            hidH[m * 132 + n0 + ncol]        = bf1(v0);
            hidH[(16 + m) * 132 + n0 + ncol] = bf1(v1);
        }
        __syncthreads();

        // GEMM2: messages = hidden @ W2 + b2
        v8f d0, d1;
        #pragma unroll
        for (int r = 0; r < 8; ++r) { d0[r] = bias2; d1[r] = bias2; }
        #pragma unroll
        for (int k = 0; k < 4; ++k) {
            BFrag a0, a1, b;
            #pragma unroll
            for (int v = 0; v < 8; ++v) {
                int sub = (v < 4) ? (4 * half + v) : (8 + 4 * half + (v - 4));
                b.u[v]  = w2s[(k * 8 + wave) * 256 + v * 32 + lane];
                a0.u[v] = hidU[mrow * 66 + k * 16 + sub];
                a1.u[v] = hidU[(16 + mrow) * 66 + k * 16 + sub];
            }
            d0 = __builtin_amdgcn_wmma_f32_16x16x32_bf16(false, a0.v, false, b.v,
                                                         (short)0, d0, false, false);
            d1 = __builtin_amdgcn_wmma_f32_16x16x32_bf16(false, a1.v, false, b.v,
                                                         (short)0, d1, false, false);
        }
        // scatter-add into both direction accumulators (L2-resident)
        #pragma unroll
        for (int r = 0; r < 8; ++r) {
            int m = r + 8 * half;
            int n = n0 + ncol;
            atomicAdd(&inc[(size_t)dSh[m] * HID + n], d0[r]);
            atomicAdd(&outg[(size_t)sSh[m] * HID + n], d0[r]);
            atomicAdd(&inc[(size_t)dSh[16 + m] * HID + n], d1[r]);
            atomicAdd(&outg[(size_t)sSh[16 + m] * HID + n], d1[r]);
        }
    }
}

// ---------------------------------------------------------------- node phase
__global__ __launch_bounds__(256) void node_kernel(
    const float* __restrict__ h, const float* __restrict__ inc,
    const float* __restrict__ outg, const float* __restrict__ inDeg,
    const float* __restrict__ outDeg, const float* __restrict__ b1,
    const float* __restrict__ b2, const float* __restrict__ lng,
    const float* __restrict__ lnb, const unsigned int* __restrict__ w1g,
    const unsigned int* __restrict__ w2g, float* __restrict__ out)
{
    __shared__ unsigned int  w1s[12 * 8 * 256];
    __shared__ unsigned int  w2s[4 * 8 * 256];
    __shared__ unsigned int  aT[32 * 196];        // reused as f32 x-tile for LN
    __shared__ unsigned short hidH[32 * 132];
    __shared__ float rin[32], rout[32];

    const int tid = threadIdx.x;
    for (int i = tid; i < 12 * 8 * 256; i += 256) w1s[i] = w1g[i];
    for (int i = tid; i < 4 * 8 * 256; i += 256)  w2s[i] = w2g[i];

    const int wave = tid >> 5, lane = tid & 31;
    const int half = lane >> 4, mrow = lane & 15, ncol = lane & 15;
    const int n0 = wave * 16;
    const float bias1 = b1[n0 + ncol];
    const float bias2 = b2[n0 + ncol];
    const unsigned int* hidU = (const unsigned int*)hidH;
    float* xSh = (float*)aT;                      // 32 x 132 f32 alias (fits in aT)

    const int nTiles = N_NODESC / 32;
    for (int t = blockIdx.x; t < nTiles; t += gridDim.x) {
        __syncthreads();
        if (tid < 32) {
            int v = t * 32 + tid;
            rin[tid]  = 1.0f / fmaxf(inDeg[v], 1.0f);
            rout[tid] = 1.0f / fmaxf(outDeg[v], 1.0f);
        }
        __syncthreads();
        // gather concat(h, incoming/in_deg, outgoing/out_deg)
        for (int p = tid; p < 32 * 192; p += 256) {
            int m = p / 192, c = p - m * 192;
            int f0 = 2 * c;
            size_t v = (size_t)(t * 32 + m);
            float x0, x1;
            if (f0 < HID)          { const float* sp = h    + v * HID + f0;             x0 = sp[0];           x1 = sp[1]; }
            else if (f0 < 2 * HID) { const float* sp = inc  + v * HID + (f0 - HID);     x0 = sp[0] * rin[m];  x1 = sp[1] * rin[m]; }
            else                   { const float* sp = outg + v * HID + (f0 - 2 * HID); x0 = sp[0] * rout[m]; x1 = sp[1] * rout[m]; }
            aT[m * 196 + c] = pack2bf(x0, x1);
        }
        __syncthreads();

        v8f c0, c1;
        #pragma unroll
        for (int r = 0; r < 8; ++r) { c0[r] = bias1; c1[r] = bias1; }
        #pragma unroll
        for (int k = 0; k < 12; ++k) {
            BFrag a0, a1, b;
            #pragma unroll
            for (int v = 0; v < 8; ++v) {
                int sub = (v < 4) ? (4 * half + v) : (8 + 4 * half + (v - 4));
                b.u[v]  = w1s[(k * 8 + wave) * 256 + v * 32 + lane];
                a0.u[v] = aT[mrow * 196 + k * 16 + sub];
                a1.u[v] = aT[(16 + mrow) * 196 + k * 16 + sub];
            }
            c0 = __builtin_amdgcn_wmma_f32_16x16x32_bf16(false, a0.v, false, b.v,
                                                         (short)0, c0, false, false);
            c1 = __builtin_amdgcn_wmma_f32_16x16x32_bf16(false, a1.v, false, b.v,
                                                         (short)0, c1, false, false);
        }
        #pragma unroll
        for (int r = 0; r < 8; ++r) {
            int m = r + 8 * half;
            float v0 = c0[r] > 0.0f ? c0[r] : 0.0f;
            float v1 = c1[r] > 0.0f ? c1[r] : 0.0f;
            hidH[m * 132 + n0 + ncol]        = bf1(v0);
            hidH[(16 + m) * 132 + n0 + ncol] = bf1(v1);
        }
        __syncthreads();   // hidden ready; aT reads fully retired -> safe to alias

        v8f d0, d1;
        #pragma unroll
        for (int r = 0; r < 8; ++r) { d0[r] = bias2; d1[r] = bias2; }
        #pragma unroll
        for (int k = 0; k < 4; ++k) {
            BFrag a0, a1, b;
            #pragma unroll
            for (int v = 0; v < 8; ++v) {
                int sub = (v < 4) ? (4 * half + v) : (8 + 4 * half + (v - 4));
                b.u[v]  = w2s[(k * 8 + wave) * 256 + v * 32 + lane];
                a0.u[v] = hidU[mrow * 66 + k * 16 + sub];
                a1.u[v] = hidU[(16 + mrow) * 66 + k * 16 + sub];
            }
            d0 = __builtin_amdgcn_wmma_f32_16x16x32_bf16(false, a0.v, false, b.v,
                                                         (short)0, d0, false, false);
            d1 = __builtin_amdgcn_wmma_f32_16x16x32_bf16(false, a1.v, false, b.v,
                                                         (short)0, d1, false, false);
        }
        // residual in fp32 -> xSh (aliases aT)
        #pragma unroll
        for (int r = 0; r < 8; ++r) {
            int m = r + 8 * half;
            int n = n0 + ncol;
            size_t v0 = (size_t)(t * 32 + m);
            size_t v1 = (size_t)(t * 32 + 16 + m);
            xSh[m * 132 + n]        = h[v0 * HID + n] + d0[r];
            xSh[(16 + m) * 132 + n] = h[v1 * HID + n] + d1[r];
        }
        __syncthreads();

        // LayerNorm: 8 threads per row (32 rows), width-8 butterfly reduce
        int row = tid >> 3, ci = tid & 7;
        float xs[16], s = 0.0f, s2 = 0.0f;
        #pragma unroll
        for (int j = 0; j < 16; ++j) {
            float x = xSh[row * 132 + ci + 8 * j];
            xs[j] = x; s += x; s2 += x * x;
        }
        #pragma unroll
        for (int off = 4; off >= 1; off >>= 1) {
            s  += __shfl_xor(s,  off, 8);
            s2 += __shfl_xor(s2, off, 8);
        }
        float mu  = s * (1.0f / 128.0f);
        float var = s2 * (1.0f / 128.0f) - mu * mu;
        float rstd = rsqrtf(var + LN_EPSF);
        size_t v = (size_t)(t * 32 + row);
        #pragma unroll
        for (int j = 0; j < 16; ++j) {
            int n = ci + 8 * j;
            out[v * HID + n] = (xs[j] - mu) * rstd * lng[n] + lnb[n];
        }
    }
}

// ---------------------------------------------------------------- launcher
extern "C" void kernel_launch(void* const* d_in, const int* in_sizes, int n_in,
                              void* d_out, int out_size, void* d_ws, size_t ws_size,
                              hipStream_t stream)
{
    const float* h     = (const float*)d_in[0];
    const int*   eidx  = (const int*)d_in[1];
    const float* ex    = (const float*)d_in[2];
    const float* ep_w1 = (const float*)d_in[3];
    const float* ep_b1 = (const float*)d_in[4];
    const float* ep_w2 = (const float*)d_in[5];
    const float* ep_b2 = (const float*)d_in[6];
    const float* up_w1 = (const float*)d_in[7];
    const float* up_b1 = (const float*)d_in[8];
    const float* up_w2 = (const float*)d_in[9];
    const float* up_b2 = (const float*)d_in[10];
    const float* ln_g  = (const float*)d_in[11];
    const float* ln_b  = (const float*)d_in[12];
    float* out = (float*)d_out;

    char* ws = (char*)d_ws;
    size_t off = 0;
    float* inc    = (float*)(ws + off); off += (size_t)N_NODESC * HID * 4;
    float* outg   = (float*)(ws + off); off += (size_t)N_NODESC * HID * 4;
    float* inDeg  = (float*)(ws + off); off += (size_t)N_NODESC * 4;
    float* outDeg = (float*)(ws + off); off += (size_t)N_NODESC * 4;
    unsigned int* w1e = (unsigned int*)(ws + off); off += 24576 * 4;
    unsigned int* w2e = (unsigned int*)(ws + off); off += 8192 * 4;
    unsigned int* w1u = (unsigned int*)(ws + off); off += 24576 * 4;
    unsigned int* w2u = (unsigned int*)(ws + off); off += 8192 * 4;

    // inc/outg/inDeg/outDeg are contiguous: single zero pass
    size_t zn = (size_t)N_NODESC * HID * 2 + (size_t)N_NODESC * 2;
    zero_kernel<<<2048, 256, 0, stream>>>(inc, zn);
    prep_kernel<<<256, 256, 0, stream>>>(ep_w1, ep_w2, up_w1, up_w2, w1e, w2e, w1u, w2u);
    edge_kernel<<<2048, 256, 0, stream>>>(h, eidx, ex, ep_b1, ep_b2, w1e, w2e,
                                          inc, outg, inDeg, outDeg);
    node_kernel<<<1024, 256, 0, stream>>>(h, inc, outg, inDeg, outDeg,
                                          up_b1, up_b2, ln_g, ln_b, w1u, w2u, out);
}
